// LSTMDecoder_90778428768308
// MI455X (gfx1250) — compile-verified, matrix operand order checked by reading, not run on hardware
//
#include <hip/hip_runtime.h>
#include <hip/hip_bf16.h>
#include <math.h>

// ---- problem constants ----
#define BB 128      // batch
#define NN 2048     // attention positions
#define TT 128      // timesteps
#define VV 80       // vocab
#define EE 128      // encoder dim
#define DD 200      // hidden dim
#define AA 128      // attention dim
#define KC 416      // padded concat K  (80 + 128 + 200 + 8)
#define KD 224      // padded D (200 -> 224)
#define G4 800      // 4*D gates

typedef _Float16 h16;
typedef __attribute__((ext_vector_type(16))) _Float16 v16h;
typedef __attribute__((ext_vector_type(8)))  float    v8f;

// ---------------- WMMA helpers (CDNA5 16x16x32 f16) ----------------
// A-frag: lane holds row r0+(lane&15); K pairs per VGPR per ISA 7.12.2.
// B-frag is loaded from a PRE-TRANSPOSED matrix (rows = N, cols = K),
// making its layout identical to an A-frag load. Produces two contiguous
// 16B loads per fragment -> global_load_b128 pairs.
__device__ __forceinline__ v16h frag_ld(const h16* p, int ld, int r0, int k0, int lane) {
  int r = r0 + (lane & 15);
  int g = lane >> 4;
  const h16* base = p + (size_t)r * ld + k0 + 8 * g;
  v16h f;
#pragma unroll
  for (int q = 0; q < 8; ++q) {
    int k = (q < 4) ? (2 * q) : (16 + 2 * (q - 4));
    f[2 * q]     = base[k];
    f[2 * q + 1] = base[k + 1];
  }
  return f;
}

__device__ __forceinline__ v8f wmma32(v16h a, v16h b, v8f c) {
  return __builtin_amdgcn_wmma_f32_16x16x32_f16(false, a, false, b, (short)0, c, false, false);
}

__device__ __forceinline__ float sigm(float x) { return 1.f / (1.f + expf(-x)); }

// ---------------- P0: stable descending argsort of lengths ----------------
__global__ void k_sort(const int* lens, int* order, int* lens_sorted) {
  int b = threadIdx.x;
  if (b >= BB) return;
  int lb = lens[b];
  int r = 0;
  for (int i = 0; i < BB; ++i) {
    int li = lens[i];
    r += (li > lb) || (li == lb && i < b);
  }
  order[r] = b;
  lens_sorted[r] = lb;
}

// ---------------- P1: gather sorted encoder + convert to f16 ----------------
// NT loads: the fp32 source is touched once; keep it out of L2 so the f16
// working set (att1 + encS, 134MB vs 192MB L2) stays resident.
__global__ void k_gather_enc(const float* enc, const int* order, h16* encS) {
  size_t idx = (size_t)blockIdx.x * blockDim.x + threadIdx.x;
  const size_t per_b = (size_t)NN * EE;
  if (idx >= (size_t)BB * per_b) return;
  size_t b = idx / per_b, rem = idx % per_b;
  encS[idx] = (h16)__builtin_nontemporal_load(enc + (size_t)order[b] * per_b + rem);
}

// ---------------- P2: mean over N (fp32, from original data) ----------------
__global__ void k_mean(const float* enc, const int* order, float* mean) {
  int b = blockIdx.x, e = threadIdx.x;   // block 128
  const float* p = enc + (size_t)order[b] * NN * EE + e;
  float s = 0.f;
  for (int n = 0; n < NN; ++n) s += __builtin_nontemporal_load(p + (size_t)n * EE);
  mean[b * EE + e] = s * (1.f / NN);
}

// ---------------- P3: h0 = mean@Wh0+bh0 ; c0 = mean@Wc0+bc0 ----------------
__global__ void k_init_hc(const float* mean, const float* Wh0, const float* bh0,
                          const float* Wc0, const float* bc0, h16* hf16, float* c) {
  int b = blockIdx.x, d = threadIdx.x;   // block 256
  if (d < DD) {
    float h = bh0[d], cc = bc0[d];
    for (int e = 0; e < EE; ++e) {
      float m = mean[b * EE + e];
      h += m * Wh0[e * DD + d];
      cc += m * Wc0[e * DD + d];
    }
    c[b * DD + d] = cc;
    hf16[b * KD + d] = (h16)h;
  } else if (d < KD) {
    hf16[b * KD + d] = (h16)0.f;         // zero pad region
  }
}

// ---------------- P4: weight prep (transpose + pad + f16) ----------------
__global__ void k_prep_w(const float* We, const float* Wd, const float* Wih,
                         const float* bih, const float* Whh, const float* bhh,
                         h16* WeT, h16* WdT, h16* WcatT, float* bcat) {
  int idx = blockIdx.x * blockDim.x + threadIdx.x;
  if (idx < G4 * KC) {                          // WcatT[j][k], j<800, k<416
    int j = idx / KC, k = idx % KC;
    float v = (k < VV + EE) ? Wih[(size_t)k * G4 + j]
            : (k < VV + EE + DD) ? Whh[(size_t)(k - VV - EE) * G4 + j] : 0.f;
    WcatT[idx] = (h16)v;
  }
  if (idx < G4) bcat[idx] = bih[idx] + bhh[idx];
  if (idx < AA * KD) {                          // WdT[a][k], k<224 (pad 200..223)
    int a = idx / KD, k = idx % KD;
    WdT[idx] = (h16)((k < DD) ? Wd[(size_t)k * AA + a] : 0.f);
  }
  if (idx < AA * EE) {                          // WeT[a][e]
    int a = idx / EE, e = idx % EE;
    WeT[idx] = (h16)We[(size_t)e * AA + a];
  }
}

// ---------------- P5: att1 = encS @ We + be  (262144x128 @ 128x128, WMMA) ----------------
__global__ void k_att1(const h16* encS, const h16* WeT, const float* be, h16* att1) {
  int lane = threadIdx.x & 31;
  int wid  = threadIdx.x >> 5;   // 8 waves per block
  int m0 = blockIdx.x * 16;      // grid 16384
  int n0 = wid * 16;
  v8f acc = {};
  for (int k0 = 0; k0 < EE; k0 += 32)
    acc = wmma32(frag_ld(encS, EE, m0, k0, lane), frag_ld(WeT, EE, n0, k0, lane), acc);
  int n = n0 + (lane & 15), g = lane >> 4;
#pragma unroll
  for (int i = 0; i < 8; ++i) {
    int m = m0 + i + 8 * g;
    att1[(size_t)m * AA + n] = (h16)(acc[i] + be[n]);
  }
}

// ---------------- S1: att2 + scores + softmax + context + xcat (fused, per-b block) ----------------
__global__ void k_attn(const h16* att1, const h16* WdT, const float* bd,
                       const float* wf, const float* bf, const h16* encS,
                       const h16* hf16, const float* tgt, const int* order,
                       const int* lens_sorted, h16* xcat, float* alphas_out, int t) {
  __shared__ float sE[NN];         // scores -> exp -> alpha
  __shared__ float sAtt2[AA];
  __shared__ float sWf[AA];
  __shared__ float sHrow[KD];
  __shared__ float sRed[256];
  __shared__ float sCtx[EE * 8];   // context partials (8 n-groups)
  int b = blockIdx.x, tid = threadIdx.x;   // block 256
  float* aout = alphas_out + ((size_t)b * TT + t) * NN;
  bool active = t < lens_sorted[b];
  if (!active) {                            // masked row: zero alphas, keep h,c
    for (int n = tid; n < NN; n += 256)
      __builtin_nontemporal_store(0.f, aout + n);
    return;
  }

  h16* xr = xcat + (size_t)b * KC;
  if (tid < AA) sWf[tid] = wf[tid];
  if (tid < KD) sHrow[tid] = (float)hf16[b * KD + tid];
  if (tid < VV)                            // x_t part of xcat
    xr[tid] = (h16)__builtin_nontemporal_load(
        tgt + ((size_t)order[b] * TT + t) * VV + tid);
  for (int j = tid; j < DD + 8; j += 256)  // h + zero pad part of xcat
    xr[VV + EE + j] = (j < DD) ? hf16[b * KD + j] : (h16)0.f;
  __syncthreads();

  // phase 0: att2_a = h . Wd[:,a] + bd  (tiny, VALU; WdT rows contiguous)
  if (tid < AA) {
    const h16* wr = WdT + (size_t)tid * KD;
    float acc = bd[tid];
#pragma unroll 8
    for (int k = 0; k < KD; ++k) acc += sHrow[k] * (float)wr[k];
    sAtt2[tid] = acc;
  }
  __syncthreads();

  // phase A: scores e_n = wf . relu(att1 + att2) + bf
  float lmax = -3.4e38f;
  for (int n = tid; n < NN; n += 256) {
    const h16* r = att1 + ((size_t)b * NN + n) * AA;
    float e = bf[0];
#pragma unroll 4
    for (int a = 0; a < AA; ++a)
      e += fmaxf((float)r[a] + sAtt2[a], 0.f) * sWf[a];
    sE[n] = e;
    lmax = fmaxf(lmax, e);
  }
  sRed[tid] = lmax; __syncthreads();
  for (int s = 128; s > 0; s >>= 1) { if (tid < s) sRed[tid] = fmaxf(sRed[tid], sRed[tid + s]); __syncthreads(); }
  float m = sRed[0]; __syncthreads();

  // phase B: exp + sum
  float lsum = 0.f;
  for (int n = tid; n < NN; n += 256) { float v = expf(sE[n] - m); sE[n] = v; lsum += v; }
  sRed[tid] = lsum; __syncthreads();
  for (int s = 128; s > 0; s >>= 1) { if (tid < s) sRed[tid] += sRed[tid + s]; __syncthreads(); }
  float inv = 1.f / sRed[0]; __syncthreads();

  // phase C: alpha out (NT store: 134MB stream, keep out of L2)
  for (int n = tid; n < NN; n += 256) {
    float a = sE[n] * inv; sE[n] = a;
    __builtin_nontemporal_store(a, aout + n);
  }
  __syncthreads();

  // phase D: awe_e = sum_n alpha_n * enc[b,n,e]; 4 e's/thread -> b64 loads
  {
    int g = tid >> 5;                 // 8 groups of 256 n
    int e4 = (tid & 31) * 4;
    const h16* eb = encS + ((size_t)b * NN + g * 256) * EE + e4;
    float a0 = 0.f, a1 = 0.f, a2 = 0.f, a3 = 0.f;
    for (int n = 0; n < 256; ++n) {
      float al = sE[g * 256 + n];
      const h16* p = eb + (size_t)n * EE;
      a0 += al * (float)p[0]; a1 += al * (float)p[1];
      a2 += al * (float)p[2]; a3 += al * (float)p[3];
    }
    sCtx[(e4 + 0) * 8 + g] = a0; sCtx[(e4 + 1) * 8 + g] = a1;
    sCtx[(e4 + 2) * 8 + g] = a2; sCtx[(e4 + 3) * 8 + g] = a3;
  }
  __syncthreads();
  if (tid < EE) {
    float s = 0.f;
#pragma unroll
    for (int g = 0; g < 8; ++g) s += sCtx[tid * 8 + g];
    xr[VV + tid] = (h16)s;            // awe part of xcat
  }
}

// ---------------- S2: gates = xcat @ [Wih;Whh] + (bih+bhh)  (128x800, K=416, WMMA) ----------------
__global__ void k_gates(const h16* xcat, const h16* WcatT, const float* bcat, float* gates) {
  int lane = threadIdx.x & 31;
  int w = blockIdx.x * 8 + (threadIdx.x >> 5);   // grid 50 -> 400 waves
  int m0 = (w / 50) * 16, n0 = (w % 50) * 16;
  v8f acc = {};
  for (int k0 = 0; k0 < KC; k0 += 32)
    acc = wmma32(frag_ld(xcat, KC, m0, k0, lane), frag_ld(WcatT, KC, n0, k0, lane), acc);
  int n = n0 + (lane & 15), g = lane >> 4;
#pragma unroll
  for (int i = 0; i < 8; ++i) {
    int m = m0 + i + 8 * g;
    gates[(size_t)m * G4 + n] = acc[i] + bcat[n];
  }
}

// ---------------- S3: LSTM cell + pred softmax (fused, per-b block) ----------------
__global__ void k_cellpred(const float* gates, float* c, h16* hf16,
                           const float* Wfc, const float* bfc,
                           const int* lens_sorted, float* preds, int t) {
  __shared__ float sH[KD];
  __shared__ float sP[256];
  int b = blockIdx.x, tid = threadIdx.x;   // block 256
  float* pout = preds + ((size_t)b * TT + t) * VV;
  bool active = t < lens_sorted[b];
  if (!active) {                           // masked: keep h,c; zero pred
    if (tid < VV) __builtin_nontemporal_store(0.f, pout + tid);
    return;
  }

  // cell
  if (tid < DD) {
    const float* gr = gates + (size_t)b * G4;
    float ig = sigm(gr[tid]);
    float fg = sigm(gr[DD + tid]);
    float gg = tanhf(gr[2 * DD + tid]);
    float og = sigm(gr[3 * DD + tid]);
    float cn = fg * c[b * DD + tid] + ig * gg;
    float hn = og * tanhf(cn);
    c[b * DD + tid] = cn;
    hf16[b * KD + tid] = (h16)hn;
    sH[tid] = hn;
  }
  __syncthreads();

  // pred = softmax(h_new @ Wfc + bfc) over V=80
  float logit = -3.4e38f;
  if (tid < VV) {
    logit = bfc[tid];
    for (int d = 0; d < DD; ++d)
      logit += sH[d] * Wfc[d * VV + tid];
  }
  sP[tid] = logit; __syncthreads();
  for (int s = 128; s > 0; s >>= 1) { if (tid < s) sP[tid] = fmaxf(sP[tid], sP[tid + s]); __syncthreads(); }
  float m = sP[0]; __syncthreads();
  float ev = (tid < VV) ? expf(logit - m) : 0.f;
  sP[tid] = ev; __syncthreads();
  for (int s = 128; s > 0; s >>= 1) { if (tid < s) sP[tid] += sP[tid + s]; __syncthreads(); }
  float inv = 1.f / sP[0];
  if (tid < VV) __builtin_nontemporal_store(ev * inv, pout + tid);
}

// ---------------- host-side launch ----------------
extern "C" void kernel_launch(void* const* d_in, const int* in_sizes, int n_in,
                              void* d_out, int out_size, void* d_ws, size_t ws_size,
                              hipStream_t stream) {
  const float* enc  = (const float*)d_in[0];
  const int*   lens = (const int*)  d_in[1];
  const float* tgt  = (const float*)d_in[2];
  const float* We   = (const float*)d_in[4];
  const float* be   = (const float*)d_in[5];
  const float* Wd   = (const float*)d_in[6];
  const float* bd   = (const float*)d_in[7];
  const float* wf   = (const float*)d_in[8];
  const float* bf   = (const float*)d_in[9];
  const float* Wih  = (const float*)d_in[10];
  const float* bih  = (const float*)d_in[11];
  const float* Whh  = (const float*)d_in[12];
  const float* bhh  = (const float*)d_in[13];
  const float* Wh0  = (const float*)d_in[14];
  const float* bh0  = (const float*)d_in[15];
  const float* Wc0  = (const float*)d_in[16];
  const float* bc0  = (const float*)d_in[17];
  const float* Wfc  = (const float*)d_in[18];
  const float* bfc  = (const float*)d_in[19];

  float* preds  = (float*)d_out;                              // (B,T,V)
  float* alphas = preds + (size_t)BB * TT * VV;               // (B,T,N)

  // workspace carve-up (256B aligned)
  char* w = (char*)d_ws; size_t off = 0;
  auto alloc = [&](size_t bytes) -> void* {
    off = (off + 255) & ~(size_t)255;
    void* p = w + off; off += bytes; return p;
  };
  int*   order       = (int*)  alloc(BB * sizeof(int));
  int*   lens_sorted = (int*)  alloc(BB * sizeof(int));
  h16*   encS        = (h16*)  alloc((size_t)BB * NN * EE * sizeof(h16));
  h16*   att1        = (h16*)  alloc((size_t)BB * NN * AA * sizeof(h16));
  float* mean        = (float*)alloc((size_t)BB * EE * sizeof(float));
  h16*   hf16        = (h16*)  alloc((size_t)BB * KD * sizeof(h16));
  float* cst         = (float*)alloc((size_t)BB * DD * sizeof(float));
  h16*   xcat        = (h16*)  alloc((size_t)BB * KC * sizeof(h16));
  h16*   WcatT       = (h16*)  alloc((size_t)G4 * KC * sizeof(h16));
  float* bcat        = (float*)alloc((size_t)G4 * sizeof(float));
  h16*   WdT         = (h16*)  alloc((size_t)AA * KD * sizeof(h16));
  h16*   WeT         = (h16*)  alloc((size_t)AA * EE * sizeof(h16));
  float* gates       = (float*)alloc((size_t)BB * G4 * sizeof(float));
  (void)ws_size; (void)n_in; (void)in_sizes; (void)out_size;

  // preprocessing
  k_sort<<<1, 128, 0, stream>>>(lens, order, lens_sorted);
  {
    size_t tot = (size_t)BB * NN * EE;
    k_gather_enc<<<(unsigned)((tot + 255) / 256), 256, 0, stream>>>(enc, order, encS);
  }
  k_mean<<<BB, EE, 0, stream>>>(enc, order, mean);
  k_init_hc<<<BB, 256, 0, stream>>>(mean, Wh0, bh0, Wc0, bc0, hf16, cst);
  {
    int tot = G4 * KC;
    k_prep_w<<<(tot + 255) / 256, 256, 0, stream>>>(We, Wd, Wih, bih, Whh, bhh,
                                                    WeT, WdT, WcatT, bcat);
  }
  k_att1<<<(BB * NN) / 16, 256, 0, stream>>>(encS, WeT, be, att1);

  // recurrent scan: 3 kernels per step
  for (int t = 0; t < TT; ++t) {
    k_attn    <<<BB, 256, 0, stream>>>(att1, WdT, bd, wf, bf, encS, hf16, tgt,
                                       order, lens_sorted, xcat, alphas, t);
    k_gates   <<<50, 256, 0, stream>>>(xcat, WcatT, bcat, gates);
    k_cellpred<<<BB, 256, 0, stream>>>(gates, cst, hf16, Wfc, bfc,
                                       lens_sorted, preds, t);
  }
}
